// Attention_13743895347813
// MI455X (gfx1250) — matrix operand order repaired and checked by reading, not compiled, not run on hardware
//
#include <hip/hip_runtime.h>
#include <hip/hip_bf16.h>

typedef __attribute__((ext_vector_type(16))) __bf16 v16bf;
typedef __attribute__((ext_vector_type(8)))  __bf16 v8bf;
typedef __attribute__((ext_vector_type(8)))  float  v8f;

#define DIMC 64
#define NKV  256
#define NTOK 16384
#define BATCH 8

// ---- fragment helpers (this code's contiguous k order, wave32) -----------
// A (16x32 bf16): lane l -> row m = l%16; K = 16*(l/16)+{0..15} (contiguous)
static __device__ __forceinline__ v16bf load_a16(const __bf16* rowp, int half) {
  const v8bf a0 = *(const v8bf*)(rowp + 16 * half);
  const v8bf a1 = *(const v8bf*)(rowp + 16 * half + 8);
  v16bf r;
#pragma unroll
  for (int i = 0; i < 8; ++i) { r[i] = a0[i]; r[i + 8] = a1[i]; }
  return r;
}

// B (32x16 bf16): lane l -> col n = l%16; K = 16*(l/16)+{0..15} (contiguous)
// => single 32-byte load from a [N][K] (transposed-weight) row.

static __device__ __forceinline__ v8f wmma_bf16(v16bf a, v16bf b, v8f c) {
  return __builtin_amdgcn_wmma_f32_16x16x32_bf16(false, a, false, b, (short)0, c,
                                                 false, false);
}

// ---- CDNA5 async copy: global -> LDS, 16B per lane, tracked by ASYNCcnt ----
static __device__ __forceinline__ void async_copy_b128(unsigned int lds_off,
                                                       unsigned long long gaddr) {
  asm volatile("global_load_async_to_lds_b128 %0, %1, off"
               :
               : "v"(lds_off), "v"(gaddr)
               : "memory");
}
static __device__ __forceinline__ void async_wait0() {
  asm volatile("s_wait_asynccnt 0x0" ::: "memory");
}

// ---- elementwise converts ---------------------------------------------------
__global__ void cvt_kernel(const float* __restrict__ src, __bf16* __restrict__ dst,
                           int n) {
  int i = blockIdx.x * blockDim.x + threadIdx.x;
  int stride = gridDim.x * blockDim.x;
  for (; i < n; i += stride) dst[i] = (__bf16)src[i];
}

__global__ void transpose_cvt_kernel(const float* __restrict__ W,
                                     __bf16* __restrict__ Wt, int K, int N) {
  int i = blockIdx.x * blockDim.x + threadIdx.x;
  if (i >= K * N) return;
  int k = i / N, n = i % N;
  Wt[(size_t)n * K + k] = (__bf16)W[(size_t)k * N + n];
}

// ---- generic M x 64 x 64 GEMM: out = (A @ W + bias) * scale ----------------
// A: bf16 [M][64] row-major, Wt: bf16 [64][64] = W transposed ([n][k]).
// storeMode 0: bf16 row-major [M][64]; storeMode 1: bf16 transposed per batch
// of 256 rows: out[b][n][kv] (V^T layout used by attention P@V).
__global__ __launch_bounds__(128) void gemm64_kernel(
    const __bf16* __restrict__ A, const __bf16* __restrict__ Wt,
    const float* __restrict__ bias, __bf16* __restrict__ out, float scale,
    int storeMode) {
  const int wave = threadIdx.x >> 5;
  const int lane = threadIdx.x & 31;
  const int half = lane >> 4, lm = lane & 15;
  const int m0 = blockIdx.x * 16;
  const int n0 = wave * 16;
  const __bf16* arow = A + (size_t)(m0 + lm) * DIMC;
  const __bf16* brow = Wt + (size_t)(n0 + lm) * DIMC;
  v8f acc = {};
#pragma unroll
  for (int kc = 0; kc < 2; ++kc) {
    v16bf a = load_a16(arow + kc * 32, half);
    v16bf b = *(const v16bf*)(brow + kc * 32 + half * 16);
    acc = wmma_bf16(a, b, acc);
  }
  const float bn = bias[n0 + lm];
  if (storeMode == 0) {
#pragma unroll
    for (int r = 0; r < 8; ++r) {
      int m = m0 + r + 8 * half;
      out[(size_t)m * DIMC + n0 + lm] = (__bf16)((acc[r] + bn) * scale);
    }
  } else {
#pragma unroll
    for (int r = 0; r < 8; ++r) {
      int m = m0 + r + 8 * half;
      int b_ = m >> 8, kv = m & 255;
      out[(size_t)b_ * (DIMC * NKV) + (size_t)(n0 + lm) * NKV + kv] =
          (__bf16)((acc[r] + bn) * scale);
    }
  }
}

// ---- spatial-reduction conv as implicit GEMM: M=2048, N=64, K=4096 ---------
// Row m=(b,oy,ox); K=(ky,kx,ci). Block = (b,oy), wave = n-tile, A row = ox.
__global__ __launch_bounds__(128) void srconv_kernel(
    const __bf16* __restrict__ xbf, const __bf16* __restrict__ WsrT,
    const float* __restrict__ bsr, float* __restrict__ xsraw) {
  const int wave = threadIdx.x >> 5;
  const int lane = threadIdx.x & 31;
  const int half = lane >> 4, lm = lane & 15;
  const int b = blockIdx.x >> 4, oy = blockIdx.x & 15;
  const int n0 = wave * 16;
  const __bf16* bbase = WsrT + (size_t)(n0 + lm) * 4096 + half * 16;
  v8f acc = {};
#pragma unroll 4
  for (int kc = 0; kc < 128; ++kc) {
    const int kbase = kc * 32;
    const int kxy = kbase >> 6;  // ky*8+kx
    const int cib = kbase & 63;  // 0 or 32
    const int ky = kxy >> 3, kx = kxy & 7;
    const __bf16* arow =
        xbf + ((size_t)b * NTOK + (size_t)(oy * 8 + ky) * 128 + lm * 8 + kx) * DIMC +
        cib;
    v16bf a = load_a16(arow, half);
    v16bf bf = *(const v16bf*)(bbase + kbase);
    acc = wmma_bf16(a, bf, acc);
  }
  const float bn = bsr[n0 + lm];
#pragma unroll
  for (int r = 0; r < 8; ++r) {
    int kv = oy * 16 + r + 8 * half;  // C row = ox
    xsraw[((size_t)b * NKV + kv) * DIMC + n0 + lm] = acc[r] + bn;
  }
}

// ---- LayerNorm over 64 channels, one wave per row --------------------------
__global__ __launch_bounds__(128) void ln_kernel(const float* __restrict__ xsraw,
                                                 const float* __restrict__ gamma,
                                                 const float* __restrict__ beta,
                                                 __bf16* __restrict__ xlnbf) {
  const int wave = threadIdx.x >> 5;
  const int lane = threadIdx.x & 31;
  const int row = blockIdx.x * 4 + wave;
  const float* rp = xsraw + (size_t)row * DIMC;
  float v0 = rp[lane], v1 = rp[lane + 32];
  float s = v0 + v1, s2 = v0 * v0 + v1 * v1;
#pragma unroll
  for (int off = 16; off >= 1; off >>= 1) {
    s += __shfl_xor(s, off, 32);
    s2 += __shfl_xor(s2, off, 32);
  }
  float mu = s * (1.0f / 64.0f);
  float var = s2 * (1.0f / 64.0f) - mu * mu;
  float rinv = rsqrtf(var + 1e-5f);
  xlnbf[(size_t)row * DIMC + lane] =
      (__bf16)((v0 - mu) * rinv * gamma[lane] + beta[lane]);
  xlnbf[(size_t)row * DIMC + lane + 32] =
      (__bf16)((v1 - mu) * rinv * gamma[lane + 32] + beta[lane + 32]);
}

// ---- fused attention: S = q@K^T (q pre-scaled), softmax, O=P@V, F=O@Wp+bp --
// K [256][64] and V^T [64][256] (bf16, 32KB each) are async-staged into LDS
// once per workgroup (4 waves share them). Dynamic LDS layout (bytes):
//   [0,32768)        kTile   (K, row-major [kv][d])
//   [32768,65536)    vTile   (V^T, row-major [d][kv])
//   [65536,98304)    pTile   per-wave 8KB: P [16][256] bf16
//   [98304,106496)   oTile   per-wave 2KB: O [16][64] bf16
__global__ __launch_bounds__(128) void attn_kernel(
    const __bf16* __restrict__ qbf, const __bf16* __restrict__ kbf,
    const __bf16* __restrict__ vtbf, const __bf16* __restrict__ WpT,
    const float* __restrict__ bp, float* __restrict__ out) {
  extern __shared__ __align__(32) char smem[];
  const int wave = threadIdx.x >> 5;
  const int lane = threadIdx.x & 31;
  const int tid = threadIdx.x;
  const int half = lane >> 4, lm = lane & 15;
  const int tile = blockIdx.x * 4 + wave;  // 16 queries per wave-tile
  const int b = tile >> 10;                // 1024 tiles per batch
  const int q0 = (tile & 1023) * 16;

  __bf16* kT = (__bf16*)smem;                       // 16384 elems
  __bf16* vT = (__bf16*)(smem + 32768);             // 16384 elems
  __bf16* pT = (__bf16*)(smem + 65536) + wave * (16 * NKV);
  __bf16* oT = (__bf16*)(smem + 98304) + wave * (16 * DIMC);

  // ---- async-stage K and V^T for this batch into LDS (16B/lane/issue) ----
  {
    const unsigned int lds_k = (unsigned int)(size_t)kT;
    const unsigned int lds_v = (unsigned int)(size_t)vT;
    const unsigned long long gk =
        (unsigned long long)(size_t)(kbf + (size_t)b * NKV * DIMC);
    const unsigned long long gv =
        (unsigned long long)(size_t)(vtbf + (size_t)b * DIMC * NKV);
#pragma unroll
    for (int it = 0; it < 16; ++it) {  // 128 threads * 16B * 16 = 32KB
      const unsigned int o = (unsigned int)(tid * 16 + it * 2048);
      async_copy_b128(lds_k + o, gk + o);
      async_copy_b128(lds_v + o, gv + o);
    }
    async_wait0();
  }
  __syncthreads();

  const __bf16* qrow = qbf + ((size_t)b * NTOK + q0 + lm) * DIMC;
  v16bf qa0 = load_a16(qrow, half);
  v16bf qa1 = load_a16(qrow + 32, half);

  // scores: S[16][256]; B operand = K rows [kv][d] from LDS
  v8f s[16];
#pragma unroll
  for (int nt = 0; nt < 16; ++nt) {
    const __bf16* krow = kT + (size_t)(nt * 16 + lm) * DIMC + half * 16;
    v8f c = {};
    c = wmma_bf16(qa0, *(const v16bf*)(krow), c);
    c = wmma_bf16(qa1, *(const v16bf*)(krow + 32), c);
    s[nt] = c;
  }

  // softmax per row; row m = r + 8*half lives across the 16 lanes of a half
  float inv[8];
#pragma unroll
  for (int r = 0; r < 8; ++r) {
    float mx = -3.0e38f;
#pragma unroll
    for (int nt = 0; nt < 16; ++nt) mx = fmaxf(mx, s[nt][r]);
#pragma unroll
    for (int off = 1; off < 16; off <<= 1) mx = fmaxf(mx, __shfl_xor(mx, off, 32));
    float sum = 0.f;
#pragma unroll
    for (int nt = 0; nt < 16; ++nt) {
      float e = __expf(s[nt][r] - mx);
      s[nt][r] = e;
      sum += e;
    }
#pragma unroll
    for (int off = 1; off < 16; off <<= 1) sum += __shfl_xor(sum, off, 32);
    inv[r] = 1.0f / sum;
  }

  // re-fragment P through LDS (C layout -> A layout)
#pragma unroll
  for (int nt = 0; nt < 16; ++nt)
#pragma unroll
    for (int r = 0; r < 8; ++r)
      pT[(r + 8 * half) * NKV + nt * 16 + lm] = (__bf16)(s[nt][r] * inv[r]);
  __syncthreads();

  // O = P @ V   (V^T rows are d, K=kv contiguous) — both operands from LDS
  v8f o[4] = {};
  for (int kc = 0; kc < 8; ++kc) {
    v16bf a = load_a16(&pT[lm * NKV + kc * 32], half);
#pragma unroll
    for (int dt = 0; dt < 4; ++dt) {
      v16bf bf = *(const v16bf*)(vT + (size_t)(dt * 16 + lm) * NKV + kc * 32 + half * 16);
      o[dt] = wmma_bf16(a, bf, o[dt]);
    }
  }

  // re-fragment O through LDS, then final projection
#pragma unroll
  for (int dt = 0; dt < 4; ++dt)
#pragma unroll
    for (int r = 0; r < 8; ++r)
      oT[(r + 8 * half) * DIMC + dt * 16 + lm] = (__bf16)o[dt][r];
  __syncthreads();

  v8f f[4] = {};
#pragma unroll
  for (int kc = 0; kc < 2; ++kc) {
    v16bf a = load_a16(&oT[lm * DIMC + kc * 32], half);
#pragma unroll
    for (int nt = 0; nt < 4; ++nt) {
      v16bf bf = *(const v16bf*)(WpT + (size_t)(nt * 16 + lm) * DIMC + kc * 32 + half * 16);
      f[nt] = wmma_bf16(a, bf, f[nt]);
    }
  }
#pragma unroll
  for (int nt = 0; nt < 4; ++nt) {
    float bn = bp[nt * 16 + lm];
#pragma unroll
    for (int r = 0; r < 8; ++r) {
      int m = q0 + r + 8 * half;
      out[((size_t)b * NTOK + m) * DIMC + nt * 16 + lm] = f[nt][r] + bn;
    }
  }
}

// ---- host side --------------------------------------------------------------
extern "C" void kernel_launch(void* const* d_in, const int* in_sizes, int n_in,
                              void* d_out, int out_size, void* d_ws, size_t ws_size,
                              hipStream_t stream) {
  const float* x = (const float*)d_in[0];
  const float* Wq = (const float*)d_in[1];
  const float* bq = (const float*)d_in[2];
  const float* Wk = (const float*)d_in[3];
  const float* bk = (const float*)d_in[4];
  const float* Wv = (const float*)d_in[5];
  const float* bv = (const float*)d_in[6];
  const float* Wsr = (const float*)d_in[7];
  const float* bsr = (const float*)d_in[8];
  const float* gamma = (const float*)d_in[9];
  const float* beta = (const float*)d_in[10];
  const float* Wp = (const float*)d_in[11];
  const float* bp = (const float*)d_in[12];
  float* out = (float*)d_out;

  char* ws = (char*)d_ws;
  __bf16* xbf   = (__bf16*)(ws);
  __bf16* qbf   = (__bf16*)(ws + 16777216);
  __bf16* WsrT  = (__bf16*)(ws + 33554432);
  __bf16* WqT   = (__bf16*)(ws + 34078720);
  __bf16* WkT   = (__bf16*)(ws + 34086912);
  __bf16* WvT   = (__bf16*)(ws + 34095104);
  __bf16* WpT   = (__bf16*)(ws + 34103296);
  float*  xsraw = (float*)(ws + 34111488);
  __bf16* xlnbf = (__bf16*)(ws + 34635776);
  __bf16* kbf   = (__bf16*)(ws + 34897920);
  __bf16* vtbf  = (__bf16*)(ws + 35160064);

  const int NX = BATCH * NTOK * DIMC;
  cvt_kernel<<<4096, 256, 0, stream>>>(x, xbf, NX);
  transpose_cvt_kernel<<<(64 * 64 + 255) / 256, 256, 0, stream>>>(Wq, WqT, 64, 64);
  transpose_cvt_kernel<<<(64 * 64 + 255) / 256, 256, 0, stream>>>(Wk, WkT, 64, 64);
  transpose_cvt_kernel<<<(64 * 64 + 255) / 256, 256, 0, stream>>>(Wv, WvT, 64, 64);
  transpose_cvt_kernel<<<(64 * 64 + 255) / 256, 256, 0, stream>>>(Wp, WpT, 64, 64);
  transpose_cvt_kernel<<<(4096 * 64 + 255) / 256, 256, 0, stream>>>(Wsr, WsrT, 4096, 64);

  // q = (x@Wq + bq) * (1/sqrt(64)); scale folded so S = q_scaled @ k^T
  gemm64_kernel<<<BATCH * NTOK / 16, 128, 0, stream>>>(xbf, WqT, bq, qbf, 0.125f, 0);
  srconv_kernel<<<BATCH * 16, 128, 0, stream>>>(xbf, WsrT, bsr, xsraw);
  ln_kernel<<<BATCH * NKV / 4, 128, 0, stream>>>(xsraw, gamma, beta, xlnbf);
  gemm64_kernel<<<BATCH * NKV / 16, 128, 0, stream>>>(xlnbf, WkT, bk, kbf, 1.0f, 0);
  gemm64_kernel<<<BATCH * NKV / 16, 128, 0, stream>>>(xlnbf, WvT, bv, vtbf, 1.0f, 1);

  const size_t attn_lds = 106496;  // 32KB K + 32KB V^T + 4*8KB P + 4*2KB O
  attn_kernel<<<BATCH * NTOK / 64, 128, attn_lds, stream>>>(qbf, kbf, vtbf, WpT, bp,
                                                            out);
}
